// CrossAttentionInteraction_25400436588553
// MI455X (gfx1250) — compile-verified, matrix operand order checked by reading, not run on hardware
//
#include <hip/hip_runtime.h>
#include <hip/hip_bf16.h>

typedef __bf16 bh;
typedef __attribute__((ext_vector_type(16))) __bf16 v16bf;
typedef __attribute__((ext_vector_type(8)))  float   v8f;
typedef __attribute__((ext_vector_type(4)))  int     v4i_t;

#define B_ 2
#define C_ 512
#define N_ 2304          // 48*48 tokens
#define ROWS (B_ * N_)   // 4608
#define HEADS_ 8
#define DHEAD 64
#define SCALE_ 0.125f

// ------------------------------------------------------- async LDS staging
#if defined(__gfx1250__) && __has_builtin(__builtin_amdgcn_global_load_async_to_lds_b128)
#define HAVE_ASYNC_LDS 1
#else
#define HAVE_ASYNC_LDS 0
#endif

__device__ __forceinline__ void stage16(const bh* g, bh* l) {
#if HAVE_ASYNC_LDS
    __builtin_amdgcn_global_load_async_to_lds_b128(
        (__attribute__((address_space(1))) v4i_t*)g,
        (__attribute__((address_space(3))) v4i_t*)l, 0, 0);
#else
    *(uint4*)l = *(const uint4*)g;   // sync fallback: VGPR round-trip
#endif
}

#if HAVE_ASYNC_LDS
#if __has_builtin(__builtin_amdgcn_s_wait_asynccnt)
#define ASYNC_WAIT(n) __builtin_amdgcn_s_wait_asynccnt(n)
#else
#define ASYNC_WAIT(n) asm volatile("s_wait_asynccnt " #n ::: "memory")
#endif
#else
#define ASYNC_WAIT(n) do {} while (0)
#endif

// ---------------------------------------------------------------- utilities
__global__ void f32_to_bf16_kernel(const float* __restrict__ in, bh* __restrict__ out, int n) {
    int i = blockIdx.x * blockDim.x + threadIdx.x;
    if (i < n) out[i] = (bh)in[i];
}

__global__ void mask_kernel(const float* __restrict__ fg_mask,
                            float* __restrict__ m, unsigned char* __restrict__ mb) {
    int i = blockIdx.x * blockDim.x + threadIdx.x;
    if (i >= B_ * N_) return;
    int b = i / N_, n = i % N_;
    int y = n / 48, x = n % 48;
    float v = fg_mask[(size_t)b * 384 * 384 + (size_t)(y * 8) * 384 + (x * 8)];
    v = fminf(fmaxf(v, 0.0f), 1.0f);
    m[i]  = v;
    mb[i] = (v >= 1.0f) ? 1 : 0;
}

// GroupNorm (32 groups of 16 ch over 2304 px) + split into fg/bg, transpose to
// token-major bf16 [B*N, 512].
__global__ __launch_bounds__(256) void groupnorm_split_kernel(
    const float* __restrict__ x, const float* __restrict__ gamma,
    const float* __restrict__ beta, const float* __restrict__ m,
    bh* __restrict__ fg, bh* __restrict__ bg) {
    const int g = blockIdx.x, b = blockIdx.y;
    const int tid = threadIdx.x;
    const float* xp = x + ((size_t)b * C_ + (size_t)g * 16) * N_;
    __shared__ float red1[256];
    __shared__ float red2[256];
    float s = 0.f, s2 = 0.f;
    for (int i = tid; i < 16 * N_; i += 256) { float v = xp[i]; s += v; s2 += v * v; }
    red1[tid] = s; red2[tid] = s2;
    __syncthreads();
    for (int st = 128; st > 0; st >>= 1) {
        if (tid < st) { red1[tid] += red1[tid + st]; red2[tid] += red2[tid + st]; }
        __syncthreads();
    }
    const float inv = 1.0f / (16.0f * N_);
    float mean = red1[0] * inv;
    float var  = red2[0] * inv - mean * mean;
    float rstd = rsqrtf(var + 1e-6f);
    for (int i = tid; i < 16 * N_; i += 256) {
        int cl = i / N_, px = i % N_;
        int c  = g * 16 + cl;
        float y  = (xp[i] - mean) * rstd * gamma[c] + beta[c];
        float mv = m[b * N_ + px];
        size_t o = (size_t)(b * N_ + px) * C_ + c;
        fg[o] = (bh)(y * mv);
        bg[o] = (bh)(y * (1.0f - mv));
    }
}

__global__ void blend_kernel(const float* __restrict__ Z, const float* __restrict__ m,
                             const float* __restrict__ adapter, float* __restrict__ out) {
    int i = blockIdx.x * blockDim.x + threadIdx.x;
    if (i >= B_ * C_ * N_) return;
    int b = i / (C_ * N_);
    int rem = i % (C_ * N_);
    int c = rem / N_;
    int n = rem % N_;
    float mv = m[b * N_ + n];
    out[i] = Z[(size_t)(b * N_ + n) * C_ + c] * mv + adapter[i] * (1.0f - mv);
}

// ---------------------------------------------------------------- WMMA GEMM
// Y[r, n] = sum_k A[r, k] * W[n, k] (+ bias[n]).  A: bf16 [M, lda], W: bf16 [Nc, ldw].
// 128 threads (4 waves), 128x64 block tile, each wave 32x64 (8 accumulators of
// v_wmma_f32_16x16x32_bf16). K stepped by 32 through double-buffered LDS filled
// with async global->LDS DMA (ASYNCcnt) when available.
__global__ __launch_bounds__(128) void gemm_bf16_wmma(
    const bh* __restrict__ A, int lda,
    const bh* __restrict__ W, int ldw,
    const float* __restrict__ bias,
    bh* __restrict__ Yb, float* __restrict__ Yf, int ldy, int K) {
    __shared__ bh As[2][128][40];  // row stride 80B (16B multiple)
    __shared__ bh Ws[2][64][40];
    const int tid  = threadIdx.x;
    const int wave = tid >> 5, lane = tid & 31;
    const int r0 = blockIdx.x * 128;
    const int c0 = blockIdx.y * 64;
    const int lr16  = lane & 15;
    const int kbase = (lane >> 4) << 3;
    const int wrow = tid >> 1;         // 0..63 : W staging row
    const int wk   = (tid & 1) * 16;   // 0 / 16

    v8f acc[2][4] = {};

    const int NK = K >> 5;
    auto stage = [&](int bu, int kt) {
        const int k0 = kt << 5;
        const bh* ga = A + (size_t)(r0 + tid) * lda + k0;   // one 32-elem row per thread
        stage16(ga,      &As[bu][tid][0]);
        stage16(ga + 8,  &As[bu][tid][8]);
        stage16(ga + 16, &As[bu][tid][16]);
        stage16(ga + 24, &As[bu][tid][24]);
        const bh* gw = W + (size_t)(c0 + wrow) * ldw + k0 + wk;
        stage16(gw,     &Ws[bu][wrow][wk]);
        stage16(gw + 8, &Ws[bu][wrow][wk + 8]);
    };

    stage(0, 0);
    for (int kt = 0; kt < NK; ++kt) {
        const int cur = kt & 1;
        if (kt + 1 < NK) { stage(cur ^ 1, kt + 1); ASYNC_WAIT(6); }
        else             { ASYNC_WAIT(0); }
        __syncthreads();   // cur buffer staged on all waves

        v16bf bfr[4];
#pragma unroll
        for (int t = 0; t < 4; ++t) {
            union { uint4 u[2]; v16bf v; } ub;
            ub.u[0] = *(const uint4*)&Ws[cur][t * 16 + lr16][kbase];
            ub.u[1] = *(const uint4*)&Ws[cur][t * 16 + lr16][16 + kbase];
            bfr[t] = ub.v;
        }
#pragma unroll
        for (int rh = 0; rh < 2; ++rh) {
            union { uint4 u[2]; v16bf v; } ua;
            ua.u[0] = *(const uint4*)&As[cur][wave * 32 + rh * 16 + lr16][kbase];
            ua.u[1] = *(const uint4*)&As[cur][wave * 32 + rh * 16 + lr16][16 + kbase];
            v16bf af = ua.v;
#pragma unroll
            for (int t = 0; t < 4; ++t)
                acc[rh][t] = __builtin_amdgcn_wmma_f32_16x16x32_bf16(
                    false, af, false, bfr[t], (short)0, acc[rh][t], false, false);
        }
        __syncthreads();   // all reads of cur done before it is restaged
    }

#pragma unroll
    for (int rh = 0; rh < 2; ++rh) {
        const int rbase = r0 + wave * 32 + rh * 16 + ((lane >> 4) << 3);
#pragma unroll
        for (int t = 0; t < 4; ++t) {
            int col = c0 + t * 16 + lr16;
            float bv = bias ? bias[col] : 0.0f;
#pragma unroll
            for (int j = 0; j < 8; ++j) {
                float v = acc[rh][t][j] + bv;
                size_t o = (size_t)(rbase + j) * ldy + col;
                if (Yf) Yf[o] = v;
                else    Yb[o] = (bh)v;
            }
        }
    }
}

// ---------------------------------------------------------------- flash attention
// One block (128 thr / 4 waves) handles 64 query rows of one (batch, head).
// Online softmax over M=2304 context tokens in 32-wide tiles; K staged to LDS
// via async DMA, V staged transposed so PV B-fragments are contiguous b128 loads.
__global__ __launch_bounds__(128) void attn_kernel(
    const bh* __restrict__ Q, const bh* __restrict__ K, const bh* __restrict__ V,
    bh* __restrict__ O, const unsigned char* __restrict__ maskb) {
    __shared__ bh Kt[32][72];      // [j][d], row stride 144B
    __shared__ bh Vtt[64][40];     // [d][j], row stride 80B
    __shared__ bh Pst[4][16][40];  // per-wave P staging

    const int tid = threadIdx.x, wave = tid >> 5, lane = tid & 31;
    const int lr16  = lane & 15;
    const int kbase = (lane >> 4) << 3;
    const int hc = blockIdx.y * DHEAD;
    const size_t rowbase = (size_t)blockIdx.z * N_;
    const int q0 = blockIdx.x * 64 + wave * 16;

    // Q fragments (16 rows x 64 d = two 16x32 A-fragments), direct b128 global loads
    v16bf aq[2];
    {
        const bh* qp = Q + (rowbase + q0 + lr16) * C_ + hc;
#pragma unroll
        for (int c = 0; c < 2; ++c) {
            union { uint4 u[2]; v16bf v; } ua;
            ua.u[0] = *(const uint4*)(qp + c * 32 + kbase);
            ua.u[1] = *(const uint4*)(qp + c * 32 + 16 + kbase);
            aq[c] = ua.v;
        }
    }

    float mrow[8], lsum[8];
#pragma unroll
    for (int j = 0; j < 8; ++j) { mrow[j] = -3.0e38f; lsum[j] = 0.0f; }
    v8f oacc[4] = {};

    const int sr = tid >> 2;          // 0..31 : K/V tile row
    const int sq = (tid & 3) * 16;    // d quarter

    for (int mt = 0; mt < N_; mt += 32) {
        __syncthreads();
        {   // stage K (async DMA, row-major) and V (transposed via VGPRs)
            const bh* kp = K + (rowbase + mt + sr) * C_ + hc + sq;
            stage16(kp,     &Kt[sr][sq]);
            stage16(kp + 8, &Kt[sr][sq + 8]);
            const bh* vp = V + (rowbase + mt + sr) * C_ + hc + sq;
            union { uint4 u; bh e[8]; } v0, v1;
            v0.u = *(const uint4*)(vp);
            v1.u = *(const uint4*)(vp + 8);
#pragma unroll
            for (int e = 0; e < 8; ++e) {
                Vtt[sq + e][sr]     = v0.e[e];
                Vtt[sq + 8 + e][sr] = v1.e[e];
            }
        }
        ASYNC_WAIT(0);
        __syncthreads();

        // S = Q K^T (two 16x16 tiles covering 32 context cols)
        v8f s[2];
#pragma unroll
        for (int cc = 0; cc < 2; ++cc) {
            v8f sacc = {};
#pragma unroll
            for (int c = 0; c < 2; ++c) {
                union { uint4 u[2]; v16bf v; } ub;
                ub.u[0] = *(const uint4*)&Kt[cc * 16 + lr16][c * 32 + kbase];
                ub.u[1] = *(const uint4*)&Kt[cc * 16 + lr16][c * 32 + 16 + kbase];
                sacc = __builtin_amdgcn_wmma_f32_16x16x32_bf16(
                    false, aq[c], false, ub.v, (short)0, sacc, false, false);
            }
            s[cc] = sacc;
        }
        // scale + context mask (column fixed per lane in C-layout)
#pragma unroll
        for (int cc = 0; cc < 2; ++cc) {
            bool msk = maskb[rowbase + mt + cc * 16 + lr16] != 0;
#pragma unroll
            for (int j = 0; j < 8; ++j) {
                float v = s[cc][j] * SCALE_;
                s[cc][j] = msk ? -3.0e38f : v;
            }
        }
        // online softmax; rows of a C-tile live within a 16-lane half
#pragma unroll
        for (int j = 0; j < 8; ++j) {
            float rm = fmaxf(s[0][j], s[1][j]);
#pragma unroll
            for (int off = 8; off >= 1; off >>= 1)
                rm = fmaxf(rm, __shfl_xor(rm, off, 32));
            float nm = fmaxf(mrow[j], rm);
            float a  = __expf(mrow[j] - nm);
            mrow[j]  = nm;
            float p0 = __expf(s[0][j] - nm);
            float p1 = __expf(s[1][j] - nm);
            float rs = p0 + p1;
#pragma unroll
            for (int off = 8; off >= 1; off >>= 1)
                rs += __shfl_xor(rs, off, 32);
            lsum[j] = lsum[j] * a + rs;
#pragma unroll
            for (int t = 0; t < 4; ++t) oacc[t][j] *= a;
            int prow = j + ((lane >> 4) << 3);
            Pst[wave][prow][lr16]      = (bh)p0;
            Pst[wave][prow][16 + lr16] = (bh)p1;
        }
        // O += P @ V   (P: 16x32 A-fragment via per-wave LDS restage)
        {
            union { uint4 u[2]; v16bf v; } ua;
            ua.u[0] = *(const uint4*)&Pst[wave][lr16][kbase];
            ua.u[1] = *(const uint4*)&Pst[wave][lr16][16 + kbase];
            v16bf ap = ua.v;
#pragma unroll
            for (int t = 0; t < 4; ++t) {
                union { uint4 u[2]; v16bf v; } ub;
                ub.u[0] = *(const uint4*)&Vtt[t * 16 + lr16][kbase];
                ub.u[1] = *(const uint4*)&Vtt[t * 16 + lr16][16 + kbase];
                oacc[t] = __builtin_amdgcn_wmma_f32_16x16x32_bf16(
                    false, ap, false, ub.v, (short)0, oacc[t], false, false);
            }
        }
    }
    const int orow = q0 + ((lane >> 4) << 3);
#pragma unroll
    for (int j = 0; j < 8; ++j) {
        float rinv = 1.0f / lsum[j];
#pragma unroll
        for (int t = 0; t < 4; ++t)
            O[(rowbase + orow + j) * C_ + hc + t * 16 + lr16] = (bh)(oacc[t][j] * rinv);
    }
}

// ---------------------------------------------------------------- host launcher
static char* carve(char*& p, size_t bytes) {
    char* r = p;
    p += (bytes + 255) & ~(size_t)255;
    return r;
}

extern "C" void kernel_launch(void* const* d_in, const int* in_sizes, int n_in,
                              void* d_out, int out_size, void* d_ws, size_t ws_size,
                              hipStream_t stream) {
    (void)in_sizes; (void)n_in; (void)out_size; (void)ws_size;
    const float* adapter = (const float*)d_in[0];
    const float* unet    = (const float*)d_in[1];
    const float* fgmask  = (const float*)d_in[2];
    const float* gamma   = (const float*)d_in[3];
    const float* beta    = (const float*)d_in[4];
    const float* Wsrc[8] = { (const float*)d_in[5],  (const float*)d_in[6],
                             (const float*)d_in[7],  (const float*)d_in[8],   // Wq1 Wk1 Wv1 Wo1
                             (const float*)d_in[10], (const float*)d_in[11],
                             (const float*)d_in[12], (const float*)d_in[13] };// Wq2 Wk2 Wv2 Wo2
    const float* bo1    = (const float*)d_in[9];
    const float* bo2    = (const float*)d_in[14];
    const float* fc_w   = (const float*)d_in[15];
    const float* fc_b   = (const float*)d_in[16];
    const float* conv_w = (const float*)d_in[17];
    const float* conv_b = (const float*)d_in[18];

    char* p = (char*)d_ws;
    float*         m_     = (float*)carve(p, (size_t)B_ * N_ * 4);
    unsigned char* mb_    = (unsigned char*)carve(p, (size_t)B_ * N_);
    bh* Wb[8];
    for (int i = 0; i < 8; ++i) Wb[i] = (bh*)carve(p, (size_t)C_ * C_ * 2);
    bh* fcWb   = (bh*)carve(p, (size_t)C_ * 2 * C_ * 2);
    bh* convWb = (bh*)carve(p, (size_t)C_ * C_ * 2);
    bh* fg   = (bh*)carve(p, (size_t)ROWS * C_ * 2);
    bh* bg   = (bh*)carve(p, (size_t)ROWS * C_ * 2);
    bh* Qb   = (bh*)carve(p, (size_t)ROWS * C_ * 2);
    bh* Kb   = (bh*)carve(p, (size_t)ROWS * C_ * 2);
    bh* Vb   = (bh*)carve(p, (size_t)ROWS * C_ * 2);
    bh* Ob   = (bh*)carve(p, (size_t)ROWS * C_ * 2);
    bh* Pcat = (bh*)carve(p, (size_t)ROWS * 2 * C_ * 2);
    bh* Tb   = (bh*)carve(p, (size_t)ROWS * C_ * 2);
    float* Zf = (float*)carve(p, (size_t)ROWS * C_ * 4);

    // mask + weight conversion
    mask_kernel<<<(B_ * N_ + 255) / 256, 256, 0, stream>>>(fgmask, m_, mb_);
    for (int i = 0; i < 8; ++i)
        f32_to_bf16_kernel<<<(C_ * C_ + 255) / 256, 256, 0, stream>>>(Wsrc[i], Wb[i], C_ * C_);
    f32_to_bf16_kernel<<<(2 * C_ * C_ + 255) / 256, 256, 0, stream>>>(fc_w, fcWb, 2 * C_ * C_);
    f32_to_bf16_kernel<<<(C_ * C_ + 255) / 256, 256, 0, stream>>>(conv_w, convWb, C_ * C_);

    dim3 gGemm(ROWS / 128, C_ / 64);
    dim3 gAttn(N_ / 64, HEADS_, B_);
    dim3 gGn(32, B_);

    for (int br = 0; br < 2; ++br) {
        const float* feat = br == 0 ? adapter : unet;
        const float* bo   = br == 0 ? bo1 : bo2;
        bh** W = &Wb[br * 4];  // Wq, Wk, Wv, Wo for this branch
        groupnorm_split_kernel<<<gGn, 256, 0, stream>>>(feat, gamma, beta, m_, fg, bg);
        gemm_bf16_wmma<<<gGemm, 128, 0, stream>>>(fg, C_, W[0], C_, nullptr, Qb, nullptr, C_, C_);
        gemm_bf16_wmma<<<gGemm, 128, 0, stream>>>(bg, C_, W[1], C_, nullptr, Kb, nullptr, C_, C_);
        gemm_bf16_wmma<<<gGemm, 128, 0, stream>>>(bg, C_, W[2], C_, nullptr, Vb, nullptr, C_, C_);
        attn_kernel<<<gAttn, 128, 0, stream>>>(Qb, Kb, Vb, Ob, mb_);
        // projection written straight into the channel-concat buffer (ldy = 1024)
        gemm_bf16_wmma<<<gGemm, 128, 0, stream>>>(Ob, C_, W[3], C_, bo,
                                                  Pcat + br * C_, nullptr, 2 * C_, C_);
    }
    // fc over concat (K = 1024), then 1x1 conv, token-major throughout
    gemm_bf16_wmma<<<gGemm, 128, 0, stream>>>(Pcat, 2 * C_, fcWb, 2 * C_, fc_b,
                                              Tb, nullptr, C_, 2 * C_);
    gemm_bf16_wmma<<<gGemm, 128, 0, stream>>>(Tb, C_, convWb, C_, conv_b,
                                              nullptr, Zf, C_, C_);
    blend_kernel<<<(B_ * C_ * N_ + 255) / 256, 256, 0, stream>>>(Zf, m_, adapter, (float*)d_out);
}
